// Attention_48369921687870
// MI455X (gfx1250) — compile-verified, matrix operand order checked by reading, not run on hardware
//
#include <hip/hip_runtime.h>
#include <hip/hip_bf16.h>

typedef __attribute__((ext_vector_type(16))) _Float16 v16h;
typedef __attribute__((ext_vector_type(8)))  float    v8f;

#define BN    16
#define NTOK  1025
#define NPAD  1088     /* 68 tiles of 16; 34 pairs of 32; 17 groups of 64 */
#define DM    512
#define NHEAD 8
#define DH    64
#define INNER 512

// ---------------------------------------------------------------------------
// CDNA5 async copy helpers: global -> LDS without touching VGPRs,
// tracked by ASYNCcnt (CDNA5 ISA).
// ---------------------------------------------------------------------------
__device__ __forceinline__ uint32_t lds_off(const void* p) {
  return (uint32_t)(uintptr_t)p;   // low 32 bits of generic LDS pointer
}

__device__ __forceinline__ void async_copy16(uint32_t dst_lds, const void* src) {
  asm volatile("global_load_async_to_lds_b128 %0, %1, off"
               :: "v"(dst_lds), "v"(src) : "memory");
}

template <int N>
__device__ __forceinline__ void wait_async() {
  asm volatile("s_wait_asynccnt %0" :: "n"(N) : "memory");
}

// ---------------------------------------------------------------------------
// Fragment helpers (16-bit A layout, ISA 7.12.2; B mirrored from N-major data)
// ---------------------------------------------------------------------------
__device__ __forceinline__ v16h load_frag(const _Float16* base, int ld) {
  const int lane = threadIdx.x & 31;
  const _Float16* p = base + (size_t)(lane & 15) * ld + ((lane >> 4) << 3);
  union { v16h h; uint4 q[2]; } u;
  u.q[0] = *(const uint4*)(p);
  u.q[1] = *(const uint4*)(p + 16);
  return u.h;
}

__device__ __forceinline__ v8f wmma16(v16h a, v16h b, v8f c) {
  return __builtin_amdgcn_wmma_f32_16x16x32_f16(false, a, false, b, (short)0, c,
                                                false, false);
}

// ---------------------------------------------------------------------------
// 1) LayerNorm: fp32 compact rows (for conv) + f16 rows padded to NPAD
// ---------------------------------------------------------------------------
__global__ __launch_bounds__(256) void ln_kernel(const float* __restrict__ x,
                                                 const float* __restrict__ g,
                                                 const float* __restrict__ bt,
                                                 float* __restrict__ xn32,
                                                 _Float16* __restrict__ xn16) {
  const int row   = blockIdx.x;            // 0 .. BN*NPAD-1
  const int bb    = row / NPAD;
  const int token = row % NPAD;
  const int t     = threadIdx.x;
  __shared__ float red[16];
  if (token >= NTOK) {                     // zero-pad f16 buffer
    xn16[(size_t)row * DM + t]       = (_Float16)0.f;
    xn16[(size_t)row * DM + t + 256] = (_Float16)0.f;
    return;
  }
  const float* xr = x + ((size_t)bb * NTOK + token) * DM;
  float v0 = xr[t], v1 = xr[t + 256];
  float s  = v0 + v1;
  float s2 = v0 * v0 + v1 * v1;
#pragma unroll
  for (int off = 16; off >= 1; off >>= 1) {
    s  += __shfl_xor(s,  off, 32);
    s2 += __shfl_xor(s2, off, 32);
  }
  if ((t & 31) == 0) { red[t >> 5] = s; red[8 + (t >> 5)] = s2; }
  __syncthreads();
  float ts = 0.f, ts2 = 0.f;
#pragma unroll
  for (int w = 0; w < 8; ++w) { ts += red[w]; ts2 += red[8 + w]; }
  const float mu  = ts * (1.0f / DM);
  const float var = ts2 * (1.0f / DM) - mu * mu;
  const float inv = rsqrtf(var + 1e-5f);
#pragma unroll
  for (int ii = 0; ii < 2; ++ii) {
    const int i = t + ii * 256;
    const float y = (xr[i] - mu) * inv * g[i] + bt[i];
    xn32[((size_t)bb * NTOK + token) * DM + i] = y;
    xn16[(size_t)row * DM + i] = (_Float16)y;
  }
}

// ---------------------------------------------------------------------------
// 2) Depthwise 5x5 conv, pad 2, spatial tokens only
// ---------------------------------------------------------------------------
__global__ __launch_bounds__(256) void dwconv_kernel(const float* __restrict__ xn32,
                                                     const float* __restrict__ w,
                                                     _Float16* __restrict__ out) {
  const int idx = blockIdx.x * 256 + threadIdx.x;   // BN*1024*512 threads
  const int c  = idx & 511;
  const int sp = (idx >> 9) & 1023;
  const int bb = idx >> 19;
  const int oy = sp >> 5, ox = sp & 31;
  const float* wc = w + c * 25;
  float acc = 0.f;
#pragma unroll
  for (int ky = 0; ky < 5; ++ky) {
    const int iy = oy + ky - 2;
    if (iy < 0 || iy > 31) continue;
#pragma unroll
    for (int kx = 0; kx < 5; ++kx) {
      const int ix = ox + kx - 2;
      if (ix < 0 || ix > 31) continue;
      acc += xn32[((size_t)(bb * NTOK + 1 + iy * 32 + ix)) * DM + c] * wc[ky * 5 + kx];
    }
  }
  out[(size_t)idx] = (_Float16)acc;
}

// ---------------------------------------------------------------------------
// 3) Weight fp32 (K,N) -> f16 transposed (N,K)
// ---------------------------------------------------------------------------
__global__ __launch_bounds__(256) void wtrans_kernel(const float* __restrict__ W,
                                                     _Float16* __restrict__ Wt,
                                                     int K, int N) {
  const int idx = blockIdx.x * 256 + threadIdx.x;   // over N*K
  if (idx >= N * K) return;
  const int n = idx / K, k = idx - n * K;
  Wt[idx] = (_Float16)W[(size_t)k * N + n];
}

// ---------------------------------------------------------------------------
// 4) f16 GEMM core: C[M,N] = A[M,K] * Bt[N,K]^T
//    Block = 8 waves = 128x64 tile. Both A (128x32) and B (64x32) k-tiles are
//    async-staged into double-buffered LDS; k-loop unrolled by 2 with static
//    ping-pong buffers and clamped (branch-free) refills. Steady state keeps
//    6 async ops in flight per wave; s_wait_asynccnt 3 retires the oldest tile.
// ---------------------------------------------------------------------------
__device__ __forceinline__ void gemm_core(const _Float16* __restrict__ A,
                                          const _Float16* __restrict__ Bt,
                                          int m0blk, int n0, int K,
                                          _Float16 (*abuf)[128 * 32],
                                          _Float16 (*bbuf)[64 * 32],
                                          v8f (&acc)[4]) {
  const int t = threadIdx.x;
  const int wave = t >> 5;
  // B tile: 64 rows x 4 b128 chunks -> 1 chunk/thread
  const int bn = t >> 2, bseg = t & 3;
  const _Float16* Bsrc = Bt + (size_t)(n0 + bn) * K + bseg * 8;
  const uint32_t Bd[2] = { lds_off(&bbuf[0][bn * 32 + bseg * 8]),
                           lds_off(&bbuf[1][bn * 32 + bseg * 8]) };
  // A tile: 128 rows x 4 b128 chunks -> 2 chunks/thread
  const int ar0 = t >> 2, ar1 = 64 + (t >> 2), aseg = t & 3;
  const _Float16* Asrc0 = A + (size_t)(m0blk + ar0) * K + aseg * 8;
  const _Float16* Asrc1 = A + (size_t)(m0blk + ar1) * K + aseg * 8;
  const uint32_t Ad0[2] = { lds_off(&abuf[0][ar0 * 32 + aseg * 8]),
                            lds_off(&abuf[1][ar0 * 32 + aseg * 8]) };
  const uint32_t Ad1[2] = { lds_off(&abuf[0][ar1 * 32 + aseg * 8]),
                            lds_off(&abuf[1][ar1 * 32 + aseg * 8]) };
  auto fill = [&](int buf, int k) {            // 3 async instr / wave (uniform)
    async_copy16(Bd[buf],  Bsrc  + k);
    async_copy16(Ad0[buf], Asrc0 + k);
    async_copy16(Ad1[buf], Asrc1 + k);
  };
  const _Float16* aw[2] = { &abuf[0][wave * 16 * 32], &abuf[1][wave * 16 * 32] };
  const int steps = K / 32;                    // even (K = 512 -> 16)
  fill(0, 0);
  fill(1, 32);
  for (int kt = 0; kt < steps; kt += 2) {
    wait_async<3>();                           // tile kt landed
    __syncthreads();
    {
      const v16h a = load_frag(aw[0], 32);
#pragma unroll
      for (int f = 0; f < 4; ++f)
        acc[f] = wmma16(a, load_frag(&bbuf[0][f * 16 * 32], 32), acc[f]);
    }
    __syncthreads();                           // reads done before refill
    int ka = (kt + 2) * 32; if (ka > K - 32) ka = K - 32;   // clamped refill
    fill(0, ka);
    wait_async<3>();                           // tile kt+1 landed
    __syncthreads();
    {
      const v16h a = load_frag(aw[1], 32);
#pragma unroll
      for (int f = 0; f < 4; ++f)
        acc[f] = wmma16(a, load_frag(&bbuf[1][f * 16 * 32], 32), acc[f]);
    }
    __syncthreads();
    int kb = (kt + 3) * 32; if (kb > K - 32) kb = K - 32;
    fill(1, kb);
  }
  wait_async<0>();                             // drain trailing refills
}

__global__ __launch_bounds__(256) void gemm_f16(const _Float16* __restrict__ A,
                                                const _Float16* __restrict__ Bt,
                                                _Float16* __restrict__ C,
                                                int N, int K) {
  __shared__ __align__(16) _Float16 abuf[2][128 * 32];
  __shared__ __align__(16) _Float16 bbuf[2][64 * 32];
  const int wave = threadIdx.x >> 5;
  const int m0 = blockIdx.x * 128 + wave * 16;
  const int n0 = blockIdx.y * 64;
  v8f acc[4] = {};
  gemm_core(A, Bt, blockIdx.x * 128, n0, K, abuf, bbuf, acc);
  const int lane = threadIdx.x & 31;
  const int rb = (lane >> 4) << 3, col = lane & 15;
#pragma unroll
  for (int f = 0; f < 4; ++f)
#pragma unroll
    for (int r = 0; r < 8; ++r)
      C[(size_t)(m0 + rb + r) * N + n0 + f * 16 + col] = (_Float16)acc[f][r];
}

// ---------------------------------------------------------------------------
// 5) Rotary + head split: Q,K (bh,NPAD,64) row-major; V transposed (bh,64,NPAD)
// ---------------------------------------------------------------------------
__global__ __launch_bounds__(256) void rope_split_kernel(const _Float16* __restrict__ xn16,
                                                         const _Float16* __restrict__ qsp,
                                                         const _Float16* __restrict__ kvb,
                                                         _Float16* __restrict__ Q,
                                                         _Float16* __restrict__ Km,
                                                         _Float16* __restrict__ Vt) {
  const size_t idx = (size_t)blockIdx.x * 256 + threadIdx.x; // BN*NPAD*8*32
  const int j     = (int)(idx & 31);
  const int head  = (int)((idx >> 5) & 7);
  const size_t rw = idx >> 8;              // bb*NPAD + token
  const int token = (int)(rw % NPAD);
  const int bb    = (int)(rw / NPAD);
  const int bh    = bb * NHEAD + head;
  const int d0    = 2 * j;

  float q0 = 0.f, q1 = 0.f, k0 = 0.f, k1 = 0.f, v0 = 0.f, v1 = 0.f;
  if (token < NTOK) {
    if (token == 0) {
      const _Float16* cr = xn16 + (size_t)bb * NPAD * DM + head * DH + d0;
      q0 = (float)cr[0]; q1 = (float)cr[1];
    } else {
      const _Float16* qr = qsp + ((size_t)(bb * 1024 + token - 1)) * INNER + head * DH + d0;
      q0 = (float)qr[0]; q1 = (float)qr[1];
    }
    const _Float16* kvr = kvb + ((size_t)bb * NPAD + token) * (2 * INNER);
    k0 = (float)kvr[head * DH + d0];
    k1 = (float)kvr[head * DH + d0 + 1];
    v0 = (float)kvr[INNER + head * DH + d0];
    v1 = (float)kvr[INNER + head * DH + d0 + 1];
  }
  // rotary: linspace(1, 640, 32) * pi, one angle per (2j, 2j+1) pair
  const float fr  = (1.0f + 639.0f * (j * (1.0f / 31.0f))) * 3.14159265358979f;
  const float ang = (float)token * fr;
  const float cs = cosf(ang), sn = sinf(ang);
  const float qr0 = q0 * cs - q1 * sn, qr1 = q1 * cs + q0 * sn;
  const float kr0 = k0 * cs - k1 * sn, kr1 = k1 * cs + k0 * sn;

  _Float16* qd = Q  + ((size_t)bh * NPAD + token) * DH + d0;
  _Float16* kd = Km + ((size_t)bh * NPAD + token) * DH + d0;
  qd[0] = (_Float16)qr0; qd[1] = (_Float16)qr1;
  kd[0] = (_Float16)kr0; kd[1] = (_Float16)kr1;
  Vt[((size_t)bh * DH + d0)     * NPAD + token] = (_Float16)v0;
  Vt[((size_t)bh * DH + d0 + 1) * NPAD + token] = (_Float16)v1;
}

// ---------------------------------------------------------------------------
// 6) Flash attention: 4 waves/block, each wave one 16-query tile of the same
//    (batch,head). K/V key-tiles async-staged into double-buffered LDS shared
//    by the 4 waves; key loop = 17 branch-free iterations of 64 keys.
// ---------------------------------------------------------------------------
__global__ __launch_bounds__(128) void attn_kernel(const _Float16* __restrict__ Q,
                                                   const _Float16* __restrict__ Km,
                                                   const _Float16* __restrict__ Vt,
                                                   _Float16* __restrict__ O) {
  __shared__ __align__(16) _Float16 kbuf[2][32 * 64];   // 32 keys x 64 dh
  __shared__ __align__(16) _Float16 vbuf[2][64 * 32];   // 64 dh x 32 keys
  __shared__ __align__(16) _Float16 pls[4][16 * 32];    // per-wave P staging
  const int wave  = threadIdx.x >> 5;
  const int lane  = threadIdx.x & 31;
  const int bh    = blockIdx.x;                         // 0..127
  int itile = blockIdx.y * 4 + wave;                    // 16-query tile (<=67)
  if (itile > NPAD / 16 - 1) itile = NPAD / 16 - 1;
  const int bb = bh >> 3, head = bh & 7;
  const float scale = 0.125f;                           // dh^-0.5

  const _Float16* Qb = Q + ((size_t)bh * NPAD + itile * 16) * DH;
  const v16h qa0 = load_frag(Qb, DH);                   // dh 0..31
  const v16h qa1 = load_frag(Qb + 32, DH);              // dh 32..63
  const _Float16* Kb = Km + (size_t)bh * NPAD * DH;
  const _Float16* Vb = Vt + (size_t)bh * DH * NPAD;

  // cooperative async fills: 4 async instr / wave per tile (uniform)
  const int t = threadIdx.x;
  auto fillKV = [&](int buf, int key0) {
#pragma unroll
    for (int c = 0; c < 2; ++c) {                       // K: 32 tok x 8 chunks
      const int ch = t + c * 128;
      const int tok = ch >> 3, seg = ch & 7;
      async_copy16(lds_off(&kbuf[buf][tok * 64 + seg * 8]),
                   Kb + (size_t)(key0 + tok) * DH + seg * 8);
    }
#pragma unroll
    for (int c = 0; c < 2; ++c) {                       // V: 64 dh x 4 chunks
      const int ch = t + c * 128;
      const int dh = ch >> 2, seg = ch & 3;
      async_copy16(lds_off(&vbuf[buf][dh * 32 + seg * 8]),
                   Vb + (size_t)dh * NPAD + key0 + seg * 8);
    }
  };

  v8f o[4] = {};
  float rm[8], rs[8];
#pragma unroll
  for (int r = 0; r < 8; ++r) { rm[r] = -1e30f; rs[r] = 0.f; }
  const int rb = (lane >> 4) << 3;
  const int cl = lane & 15;

  auto process = [&](const _Float16* kb, const _Float16* vb, const _Float16* pw,
                     int key0) {
    v8f s0 = {}, s1 = {};
    {
      const v16h b0  = load_frag(kb, DH);
      const v16h b0b = load_frag(kb + 32, DH);
      s0 = wmma16(qa0, b0, s0);
      s0 = wmma16(qa1, b0b, s0);
      const v16h b1  = load_frag(kb + 16 * 64, DH);
      const v16h b1b = load_frag(kb + 16 * 64 + 32, DH);
      s1 = wmma16(qa0, b1, s1);
      s1 = wmma16(qa1, b1b, s1);
    }
    const bool ok0 = (key0 + cl) < NTOK;
    const bool ok1 = (key0 + 16 + cl) < NTOK;
#pragma unroll
    for (int r = 0; r < 8; ++r) {
      s0[r] = ok0 ? s0[r] * scale : -1e30f;
      s1[r] = ok1 ? s1[r] * scale : -1e30f;
    }
#pragma unroll
    for (int r = 0; r < 8; ++r) {
      float mx = fmaxf(s0[r], s1[r]);
#pragma unroll
      for (int off = 8; off >= 1; off >>= 1) mx = fmaxf(mx, __shfl_xor(mx, off, 16));
      const float nm = fmaxf(rm[r], mx);
      const float corr = __expf(rm[r] - nm);
      rm[r] = nm;
      rs[r] *= corr;
#pragma unroll
      for (int f = 0; f < 4; ++f) o[f][r] *= corr;
      const float p0 = __expf(s0[r] - nm);
      const float p1 = __expf(s1[r] - nm);
      float sm = p0 + p1;
#pragma unroll
      for (int off = 8; off >= 1; off >>= 1) sm += __shfl_xor(sm, off, 16);
      rs[r] += sm;
      s0[r] = p0; s1[r] = p1;
    }
    // restage P (C layout) -> LDS row-major 16x32 -> reload as A fragment
    _Float16* pwm = const_cast<_Float16*>(pw);
#pragma unroll
    for (int r = 0; r < 8; ++r) {
      pwm[(r + rb) * 32 + cl]      = (_Float16)s0[r];
      pwm[(r + rb) * 32 + 16 + cl] = (_Float16)s1[r];
    }
    __syncthreads();
    const v16h pa = load_frag(pw, 32);
#pragma unroll
    for (int f = 0; f < 4; ++f) {
      const v16h bv = load_frag(vb + f * 16 * 32, 32);
      o[f] = wmma16(pa, bv, o[f]);
    }
  };

  fillKV(0, 0);
  fillKV(1, 32);
  for (int p2 = 0; p2 < NPAD / 64; ++p2) {              // 17 iterations
    const int key0 = p2 * 64;
    wait_async<4>();                                    // tile key0 landed
    __syncthreads();
    process(kbuf[0], vbuf[0], pls[wave], key0);
    __syncthreads();                                    // reads done
    int ka = key0 + 64; if (ka > NPAD - 32) ka = NPAD - 32;
    fillKV(0, ka);
    wait_async<4>();                                    // tile key0+32 landed
    __syncthreads();
    process(kbuf[1], vbuf[1], pls[wave], key0 + 32);
    __syncthreads();
    int kb2 = key0 + 96; if (kb2 > NPAD - 32) kb2 = NPAD - 32;
    fillKV(1, kb2);
  }
  wait_async<0>();
  // normalize + merge heads: O[bb][token][head*64 + d]
#pragma unroll
  for (int f = 0; f < 4; ++f)
#pragma unroll
    for (int r = 0; r < 8; ++r) {
      const int token = itile * 16 + rb + r;
      if (token < NTOK) {
        const float val = o[f][r] / rs[r];
        O[((size_t)bb * NPAD + token) * INNER + head * DH + f * 16 + cl] = (_Float16)val;
      }
    }
}

// ---------------------------------------------------------------------------
// 7) Output GEMM with bias, fp32 store, row remap NPAD -> NTOK
// ---------------------------------------------------------------------------
__global__ __launch_bounds__(256) void gemm_out(const _Float16* __restrict__ A,
                                                const _Float16* __restrict__ Bt,
                                                const float* __restrict__ bias,
                                                float* __restrict__ out,
                                                int N, int K) {
  __shared__ __align__(16) _Float16 abuf[2][128 * 32];
  __shared__ __align__(16) _Float16 bbuf[2][64 * 32];
  const int wave = threadIdx.x >> 5;
  const int m0 = blockIdx.x * 128 + wave * 16;
  const int n0 = blockIdx.y * 64;
  v8f acc[4] = {};
  gemm_core(A, Bt, blockIdx.x * 128, n0, K, abuf, bbuf, acc);
  const int lane = threadIdx.x & 31;
  const int rb = (lane >> 4) << 3, col = lane & 15;
#pragma unroll
  for (int f = 0; f < 4; ++f)
#pragma unroll
    for (int r = 0; r < 8; ++r) {
      const int mrow = m0 + rb + r;
      const int bb = mrow / NPAD, token = mrow % NPAD;
      if (token < NTOK)
        out[((size_t)bb * NTOK + token) * N + n0 + f * 16 + col] =
            acc[f][r] + bias[n0 + f * 16 + col];
    }
}

// ---------------------------------------------------------------------------
extern "C" void kernel_launch(void* const* d_in, const int* in_sizes, int n_in,
                              void* d_out, int out_size, void* d_ws, size_t ws_size,
                              hipStream_t stream) {
  const float* x    = (const float*)d_in[0];
  const float* ln_g = (const float*)d_in[1];
  const float* ln_b = (const float*)d_in[2];
  const float* dw_w = (const float*)d_in[3];
  const float* pw_w = (const float*)d_in[4];
  const float* kv_w = (const float*)d_in[5];
  const float* ow_w = (const float*)d_in[6];
  const float* ow_b = (const float*)d_in[7];
  (void)in_sizes; (void)n_in; (void)out_size; (void)ws_size;

  char* ws = (char*)d_ws;
  size_t off = 0;
  auto alloc = [&](size_t bytes) {
    void* p = ws + off;
    off = (off + bytes + 255) & ~(size_t)255;
    return p;
  };

  float*     xn32   = (float*)    alloc((size_t)BN * NTOK * DM * 4);
  _Float16*  xn16   = (_Float16*) alloc((size_t)BN * NPAD * DM * 2);
  _Float16*  conv16 = (_Float16*) alloc((size_t)BN * 1024 * DM * 2);
  _Float16*  qsp    = (_Float16*) alloc((size_t)BN * 1024 * INNER * 2);
  _Float16*  kvb    = (_Float16*) alloc((size_t)BN * NPAD * 2 * INNER * 2);
  _Float16*  Qh     = (_Float16*) alloc((size_t)BN * NHEAD * NPAD * DH * 2);
  _Float16*  Kh     = (_Float16*) alloc((size_t)BN * NHEAD * NPAD * DH * 2);
  _Float16*  Vth    = (_Float16*) alloc((size_t)BN * NHEAD * DH * NPAD * 2);
  _Float16*  attn_o = (_Float16*) alloc((size_t)BN * NPAD * INNER * 2);
  _Float16*  pw_t   = (_Float16*) alloc((size_t)DM * INNER * 2);
  _Float16*  kv_t   = (_Float16*) alloc((size_t)DM * 2 * INNER * 2);
  _Float16*  ow_t   = (_Float16*) alloc((size_t)INNER * DM * 2);

  // weights -> f16 transposed (N,K)
  wtrans_kernel<<<(DM * INNER + 255) / 256, 256, 0, stream>>>(pw_w, pw_t, DM, INNER);
  wtrans_kernel<<<(DM * 2 * INNER + 255) / 256, 256, 0, stream>>>(kv_w, kv_t, DM, 2 * INNER);
  wtrans_kernel<<<(INNER * DM + 255) / 256, 256, 0, stream>>>(ow_w, ow_t, INNER, DM);

  // layernorm (padded f16 + compact f32)
  ln_kernel<<<BN * NPAD, 256, 0, stream>>>(x, ln_g, ln_b, xn32, xn16);

  // depthwise conv
  dwconv_kernel<<<(BN * 1024 * DM) / 256, 256, 0, stream>>>(xn32, dw_w, conv16);

  // q spatial = conv @ pw_w   : M=16384, N=512, K=512
  gemm_f16<<<dim3(BN * 1024 / 128, INNER / 64), 256, 0, stream>>>(conv16, pw_t, qsp, INNER, DM);

  // kv = xn @ kv_w            : M=17408, N=1024, K=512
  gemm_f16<<<dim3(BN * NPAD / 128, (2 * INNER) / 64), 256, 0, stream>>>(xn16, kv_t, kvb, 2 * INNER, DM);

  // rotary + head split
  rope_split_kernel<<<(BN * NPAD * NHEAD * 32) / 256, 256, 0, stream>>>(xn16, qsp, kvb, Qh, Kh, Vth);

  // attention: 4 waves/block, each wave one q-tile; 65 live tiles -> 17 groups
  attn_kernel<<<dim3(BN * NHEAD, 17), 128, 0, stream>>>(Qh, Kh, Vth, attn_o);

  // out = attn_o @ out_w + out_b : M=17408, N=512, K=512, fp32 store
  gemm_out<<<dim3(BN * NPAD / 128, DM / 64), 256, 0, stream>>>(attn_o, ow_t, ow_b,
                                                               (float*)d_out, DM, DM);
}
